// AttentionBlock_33543694582519
// MI455X (gfx1250) — matrix operand order repaired and checked by reading, not compiled, not run on hardware
//
#include <hip/hip_runtime.h>
#include <hip/hip_bf16.h>
#include <stdint.h>

// Problem constants (match reference)
#define S_LEN 4096
#define HID   1024
#define NHEAD 2
#define HDIM  512
#define DSTAT 64
#define LN_EPS 1e-5f

typedef __attribute__((ext_vector_type(16))) __bf16 bf16x16;
typedef __attribute__((ext_vector_type(8)))  float  f32x8;

union FragCvt { uint4 q[2]; bf16x16 v; };
union Pack8   { __bf16 h[8]; uint4 q; };
union Pack4   { __bf16 h[4]; uint2 u; };

// ---- WMMA fragment loaders (CDNA5 16x16x32 bf16 VGPR layout, ISA 7.12.2) ----
// A (16xK tile, row-major, ld elements): lane holds row (lane&15),
// two 8-element runs at k + (lane>>4)*8 and k + 16 + (lane>>4)*8.
static __device__ __forceinline__ bf16x16
load_a_frag(const __bf16* base, int ld, int row0, int k0, int lane) {
    const __bf16* p = base + (size_t)(row0 + (lane & 15)) * ld
                           + (k0 + ((lane >> 4) << 3));
    FragCvt u;
    u.q[0] = *(const uint4*)p;
    u.q[1] = *(const uint4*)(p + 16);
    return u.v;
}
// B (Kx16 tile of an NT operand: B[k,n] = W[col0+n, k], W row-major, ld elems):
// lane holds column (lane&15) = one row of W; two 8-element runs at k + (lane>>4)*8 and k + 16 + (lane>>4)*8.
static __device__ __forceinline__ bf16x16
load_b_frag(const __bf16* base, int ld, int col0, int k0, int lane) {
    const __bf16* p = base + (size_t)(col0 + (lane & 15)) * ld
                           + (k0 + ((lane >> 4) << 3));
    FragCvt u;
    u.q[0] = *(const uint4*)p;
    u.q[1] = *(const uint4*)(p + 16);
    return u.v;
}
static __device__ __forceinline__ f32x8
wmma_bf16(bf16x16 a, bf16x16 b, f32x8 c) {
    return __builtin_amdgcn_wmma_f32_16x16x32_bf16(
        /*neg_a=*/false, a, /*neg_b=*/false, b,
        /*c_mod=*/(short)0, c, /*reuse_a=*/false, /*reuse_b=*/false);
}

// ---------------- semb = Ws @ static + bs ----------------
__global__ void k_semb(const float* __restrict__ Ws, const float* __restrict__ sd,
                       const float* __restrict__ bs, float* __restrict__ semb) {
    int i = blockIdx.x * blockDim.x + threadIdx.x;
    if (i >= HID) return;
    float acc = bs[i];
#pragma unroll 8
    for (int j = 0; j < DSTAT; ++j) acc += Ws[i * DSTAT + j] * sd[j];
    semb[i] = acc;
}

// ---------------- f32 -> bf16 converts (4 elems / thread) ----------------
__global__ void k_cvt(const float* __restrict__ in, __bf16* __restrict__ out, int n4) {
    int i = blockIdx.x * blockDim.x + threadIdx.x;
    if (i >= n4) return;
    float4 f = ((const float4*)in)[i];
    Pack4 p;
    p.h[0] = (__bf16)f.x; p.h[1] = (__bf16)f.y;
    p.h[2] = (__bf16)f.z; p.h[3] = (__bf16)f.w;
    ((uint2*)out)[i] = p.u;
}
__global__ void k_cvt_add(const float* __restrict__ in, const float* __restrict__ semb,
                          __bf16* __restrict__ out, int n4) {
    int i = blockIdx.x * blockDim.x + threadIdx.x;
    if (i >= n4) return;
    int c = (i * 4) & (HID - 1);
    float4 f = ((const float4*)in)[i];
    Pack4 p;
    p.h[0] = (__bf16)(f.x + semb[c]);
    p.h[1] = (__bf16)(f.y + semb[c + 1]);
    p.h[2] = (__bf16)(f.z + semb[c + 2]);
    p.h[3] = (__bf16)(f.w + semb[c + 3]);
    ((uint2*)out)[i] = p.u;
}

// ---------------- C[M,N] = A[M,K] @ B[N,K]^T  (bf16 in, f32 acc) ----------------
// mode 0: store bf16 row-major [M,N]
// mode 1: store bf16 transposed [N,M]      (used for V^T)
// mode 2: store f32 row-major, += resid    (output projection + residual)
// Block: 256 threads = 8 waves; block tile 64(M) x 128(N); wave tile 16x64.
// Inner loop: unroll-by-2 ping-pong register double-buffering. Each half-step
// loads the buffer two steps ahead directly (no rotation copies), so WMMAs
// overlap global-load latency without v_mov register shuffling.
template <int KD>
__global__ void __launch_bounds__(256)
k_gemm_nt(const __bf16* __restrict__ A, const __bf16* __restrict__ B,
          int M, int N, int mode,
          __bf16* __restrict__ outb, float* __restrict__ outf,
          const float* __restrict__ resid) {
    int lane = threadIdx.x & 31;
    int wid  = threadIdx.x >> 5;
    int wm = wid & 3, wn = wid >> 2;
    int rowBase = blockIdx.y * 64 + wm * 16;
    int colBase = blockIdx.x * 128 + wn * 64;

    f32x8 acc[4] = {};
    bf16x16 a0 = load_a_frag(A, KD, rowBase, 0, lane);
    bf16x16 b0[4];
#pragma unroll
    for (int nt = 0; nt < 4; ++nt)
        b0[nt] = load_b_frag(B, KD, colBase + nt * 16, 0, lane);

    for (int k = 0; k < KD; k += 64) {        // KD % 64 == 0
        // prefetch odd buffer (k+32): always valid since KD % 64 == 0
        bf16x16 a1 = load_a_frag(A, KD, rowBase, k + 32, lane);
        bf16x16 b1[4];
#pragma unroll
        for (int nt = 0; nt < 4; ++nt)
            b1[nt] = load_b_frag(B, KD, colBase + nt * 16, k + 32, lane);
        // consume even buffer
#pragma unroll
        for (int nt = 0; nt < 4; ++nt)
            acc[nt] = wmma_bf16(a0, b0[nt], acc[nt]);
        // prefetch next even buffer (k+64)
        if (k + 64 < KD) {
            a0 = load_a_frag(A, KD, rowBase, k + 64, lane);
#pragma unroll
            for (int nt = 0; nt < 4; ++nt)
                b0[nt] = load_b_frag(B, KD, colBase + nt * 16, k + 64, lane);
        }
        // consume odd buffer
#pragma unroll
        for (int nt = 0; nt < 4; ++nt)
            acc[nt] = wmma_bf16(a1, b1[nt], acc[nt]);
    }

    int r0 = rowBase + ((lane >> 4) << 3);
#pragma unroll
    for (int nt = 0; nt < 4; ++nt) {
        int col = colBase + nt * 16 + (lane & 15);
        if (mode == 0) {
#pragma unroll
            for (int r = 0; r < 8; ++r)
                outb[(size_t)(r0 + r) * N + col] = (__bf16)acc[nt][r];
        } else if (mode == 1) {
            Pack8 p;
#pragma unroll
            for (int r = 0; r < 8; ++r) p.h[r] = (__bf16)acc[nt][r];
            *(uint4*)(outb + (size_t)col * M + r0) = p.q;   // 8 consecutive rows
        } else {
#pragma unroll
            for (int r = 0; r < 8; ++r) {
                size_t idx = (size_t)(r0 + r) * N + col;
                outf[idx] = acc[nt][r] + resid[idx];
            }
        }
    }
}

// ---------------- flash attention, bf16 WMMA, online softmax ----------------
// grid = (S/16 query tiles, NH heads), block = 256 threads = 8 waves.
// Per key-block of 128: wave w computes the 16x16 logits tile for keys
// [kb*128 + w*16, +16) (16 WMMAs over HD=512), then all waves share St/Ps via
// LDS; wave w owns ctx columns [w*64, w*64+64) (4 accumulator tiles).
__global__ void __launch_bounds__(256)
k_attn(const __bf16* __restrict__ Q, const __bf16* __restrict__ Kb,
       const __bf16* __restrict__ Vt, const unsigned char* __restrict__ mask,
       __bf16* __restrict__ C) {
    __shared__ __align__(16) __bf16 Qs[16 * 512];   // 16 KB query tile
    __shared__ __align__(16) float  St[16 * 128];   //  8 KB logits
    __shared__ __align__(16) __bf16 Ps[16 * 128];   //  4 KB probs (bf16)
    __shared__ float mrow[16], lrow[16], arow[16];
    __shared__ float pred[16 * 8];                  // partial max/sum scratch

    int tid  = threadIdx.x;
    int lane = tid & 31;
    int wave = tid >> 5;
    int h    = blockIdx.y;
    int q0   = blockIdx.x * 16;

    // stage 16x512 Q tile into LDS (64 B per thread) via CDNA5 async DMA.
    // INST_OFFSET applies to both global and LDS sides (ISA 10 "async"),
    // and our LDS layout mirrors global within each 64 B chunk.
    {
        int row = tid >> 4, seg = tid & 15;
        const __bf16* g = Q + (size_t)(q0 + row) * HID + h * HDIM + seg * 32;
        uint32_t l = (uint32_t)(uintptr_t)(Qs + row * 512 + seg * 32);
        asm volatile("global_load_async_to_lds_b128 %0, %1, off"
                     :: "v"(l), "v"(g) : "memory");
        asm volatile("global_load_async_to_lds_b128 %0, %1, off offset:16"
                     :: "v"(l), "v"(g) : "memory");
        asm volatile("global_load_async_to_lds_b128 %0, %1, off offset:32"
                     :: "v"(l), "v"(g) : "memory");
        asm volatile("global_load_async_to_lds_b128 %0, %1, off offset:48"
                     :: "v"(l), "v"(g) : "memory");
    }
    if (tid < 16) { mrow[tid] = -3.4028235e38f; lrow[tid] = 0.f; }
    f32x8 acc[4] = {};
    asm volatile("s_wait_asynccnt 0x0" ::: "memory");
    __syncthreads();

    const float scale = 0.044194173824159216f;   // 1/sqrt(512)
    for (int kb = 0; kb < S_LEN / 128; ++kb) {
        int key0 = kb * 128 + wave * 16;
        // ---- stage 1: this wave's 16x16 logits tile (Q @ K^T), pipelined ----
        f32x8 s = {};
        {
            const __bf16* kbase = Kb + h * HDIM;
            bf16x16 bcur = load_b_frag(kbase, HID, key0, 0, lane);
#pragma unroll
            for (int kc = 0; kc < 16; ++kc) {
                bf16x16 bnxt = bcur;
                if (kc < 15)
                    bnxt = load_b_frag(kbase, HID, key0, (kc + 1) * 32, lane);
                bf16x16 a = load_a_frag(Qs, 512, 0, kc * 32, lane);
                s = wmma_bf16(a, bcur, s);
                bcur = bnxt;
            }
        }
        int rloc = (lane >> 4) << 3;                 // local row base (C layout)
        int keyCol = key0 + (lane & 15);             // C layout: lane -> column
        bool ok = (h != 0) || (mask[keyCol] != 0);   // head0: key mask; head1: all
#pragma unroll
        for (int r = 0; r < 8; ++r) {
            float v = ok ? s[r] * scale : -3.4028235e38f;  // finfo(f32).min
            St[(rloc + r) * 128 + wave * 16 + (lane & 15)] = v;
        }
        __syncthreads();
        // ---- stage 2a: partial max over 16-col strips (128 threads) ----
        if (tid < 128) {
            int row = tid >> 3, sub = tid & 7;
            const float* sp = St + row * 128 + sub * 16;
            float mx = sp[0];
#pragma unroll
            for (int j = 1; j < 16; ++j) mx = fmaxf(mx, sp[j]);
            pred[row * 8 + sub] = mx;
        }
        __syncthreads();
        // ---- stage 2b: row max + rescale factor (16 threads) ----
        if (tid < 16) {
            float mo = mrow[tid], mx = mo;
#pragma unroll
            for (int s2 = 0; s2 < 8; ++s2) mx = fmaxf(mx, pred[tid * 8 + s2]);
            arow[tid] = __expf(mo - mx);
            mrow[tid] = mx;
        }
        __syncthreads();
        // ---- stage 2c: exponentiate + partial sums (128 threads) ----
        if (tid < 128) {
            int row = tid >> 3, sub = tid & 7;
            float mx = mrow[row];
            const float* sp = St + row * 128 + sub * 16;
            __bf16* pp = Ps + row * 128 + sub * 16;
            float sum = 0.f;
#pragma unroll
            for (int j = 0; j < 16; ++j) {
                float p = __expf(sp[j] - mx);
                sum += p;
                pp[j] = (__bf16)p;
            }
            pred[row * 8 + sub] = sum;
        }
        __syncthreads();
        // ---- stage 2d: running denominator (16 threads) ----
        if (tid < 16) {
            float sum = 0.f;
#pragma unroll
            for (int s2 = 0; s2 < 8; ++s2) sum += pred[tid * 8 + s2];
            lrow[tid] = lrow[tid] * arow[tid] + sum;
        }
        __syncthreads();
        // ---- stage 3: rescale ctx accumulators, then P @ V (pipelined) ----
        float al[8];
#pragma unroll
        for (int r = 0; r < 8; ++r) al[r] = arow[rloc + r];
#pragma unroll
        for (int nt = 0; nt < 4; ++nt)
#pragma unroll
            for (int r = 0; r < 8; ++r) acc[nt][r] *= al[r];

        const __bf16* vbase = Vt + (size_t)(h * HDIM) * S_LEN;
        bf16x16 vb[4];
#pragma unroll
        for (int nt = 0; nt < 4; ++nt)
            vb[nt] = load_b_frag(vbase, S_LEN, wave * 64 + nt * 16, kb * 128, lane);
#pragma unroll
        for (int kc = 0; kc < 4; ++kc) {
            bf16x16 a = load_a_frag(Ps, 128, 0, kc * 32, lane);
            bf16x16 vn[4];
            if (kc < 3) {
#pragma unroll
                for (int nt = 0; nt < 4; ++nt)
                    vn[nt] = load_b_frag(vbase, S_LEN, wave * 64 + nt * 16,
                                         kb * 128 + (kc + 1) * 32, lane);
            }
#pragma unroll
            for (int nt = 0; nt < 4; ++nt)
                acc[nt] = wmma_bf16(a, vb[nt], acc[nt]);
            if (kc < 3) {
#pragma unroll
                for (int nt = 0; nt < 4; ++nt) vb[nt] = vn[nt];
            }
        }
    }
    __syncthreads();
    int rloc = (lane >> 4) << 3;
    float linv[8];
#pragma unroll
    for (int r = 0; r < 8; ++r) linv[r] = 1.0f / lrow[rloc + r];
#pragma unroll
    for (int nt = 0; nt < 4; ++nt) {
        int col = h * HDIM + wave * 64 + nt * 16 + (lane & 15);
#pragma unroll
        for (int r = 0; r < 8; ++r)
            C[(size_t)(q0 + rloc + r) * HID + col] = (__bf16)(acc[nt][r] * linv[r]);
    }
}

// ---------------- per-token LayerNorm (one block per row) ----------------
__global__ void __launch_bounds__(256)
k_layernorm(const float* __restrict__ res, const float* __restrict__ w,
            const float* __restrict__ b, float* __restrict__ out) {
    __shared__ float ssum[256], ssq[256];
    int row = blockIdx.x;
    float4 x = ((const float4*)(res + (size_t)row * HID))[threadIdx.x];
    ssum[threadIdx.x] = x.x + x.y + x.z + x.w;
    ssq[threadIdx.x]  = x.x * x.x + x.y * x.y + x.z * x.z + x.w * x.w;
    __syncthreads();
    for (int off = 128; off > 0; off >>= 1) {
        if (threadIdx.x < off) {
            ssum[threadIdx.x] += ssum[threadIdx.x + off];
            ssq[threadIdx.x]  += ssq[threadIdx.x + off];
        }
        __syncthreads();
    }
    float mu   = ssum[0] * (1.0f / HID);
    float var  = ssq[0] * (1.0f / HID) - mu * mu;
    float rstd = rsqrtf(var + LN_EPS);
    float4 wv = ((const float4*)w)[threadIdx.x];
    float4 bv = ((const float4*)b)[threadIdx.x];
    float4 o;
    o.x = (x.x - mu) * rstd * wv.x + bv.x;
    o.y = (x.y - mu) * rstd * wv.y + bv.y;
    o.z = (x.z - mu) * rstd * wv.z + bv.z;
    o.w = (x.w - mu) * rstd * wv.w + bv.w;
    ((float4*)(out + (size_t)row * HID))[threadIdx.x] = o;
}

extern "C" void kernel_launch(void* const* d_in, const int* in_sizes, int n_in,
                              void* d_out, int out_size, void* d_ws, size_t ws_size,
                              hipStream_t stream) {
    (void)in_sizes; (void)n_in; (void)out_size; (void)ws_size;
    const float* inputs = (const float*)d_in[0];
    const float* sdata  = (const float*)d_in[1];
    // jnp bool arrays are 1 byte per element
    const unsigned char* bmask = (const unsigned char*)d_in[2];
    const float* Wq  = (const float*)d_in[3];
    const float* Wk  = (const float*)d_in[4];
    const float* Wv  = (const float*)d_in[5];
    const float* Wo  = (const float*)d_in[6];
    const float* Ws  = (const float*)d_in[7];
    const float* bs  = (const float*)d_in[8];
    const float* lnw = (const float*)d_in[9];
    const float* lnb = (const float*)d_in[10];
    float* out = (float*)d_out;

    // workspace carve-up (256 B aligned)
    char* ws = (char*)d_ws;
    size_t off = 0;
    auto alloc = [&](size_t bytes) -> void* {
        void* p = ws + off;
        off = (off + bytes + 255) & ~(size_t)255;
        return p;
    };
    float*  semb = (float*) alloc((size_t)HID * 4);
    __bf16* Xbf  = (__bf16*)alloc((size_t)S_LEN * HID * 2);
    __bf16* XSbf = (__bf16*)alloc((size_t)S_LEN * HID * 2);
    __bf16* Wqbf = (__bf16*)alloc((size_t)HID * HID * 2);
    __bf16* Wkbf = (__bf16*)alloc((size_t)HID * HID * 2);
    __bf16* Wvbf = (__bf16*)alloc((size_t)HID * HID * 2);
    __bf16* Wobf = (__bf16*)alloc((size_t)HID * HID * 2);
    __bf16* Qbf  = (__bf16*)alloc((size_t)S_LEN * HID * 2);
    __bf16* Kbf  = (__bf16*)alloc((size_t)S_LEN * HID * 2);
    __bf16* Vt   = (__bf16*)alloc((size_t)S_LEN * HID * 2);  // [H][S]
    __bf16* Cbf  = (__bf16*)alloc((size_t)S_LEN * HID * 2);
    float*  Res  = (float*) alloc((size_t)S_LEN * HID * 4);

    // 1) static embedding
    k_semb<<<dim3(4), dim3(256), 0, stream>>>(Ws, sdata, bs, semb);

    // 2) bf16 converts
    int n4x = S_LEN * HID / 4;   // 1,048,576
    int n4w = HID * HID / 4;     //   262,144
    k_cvt    <<<dim3(n4x / 256), dim3(256), 0, stream>>>(inputs, Xbf, n4x);
    k_cvt_add<<<dim3(n4x / 256), dim3(256), 0, stream>>>(inputs, semb, XSbf, n4x);
    k_cvt    <<<dim3(n4w / 256), dim3(256), 0, stream>>>(Wq, Wqbf, n4w);
    k_cvt    <<<dim3(n4w / 256), dim3(256), 0, stream>>>(Wk, Wkbf, n4w);
    k_cvt    <<<dim3(n4w / 256), dim3(256), 0, stream>>>(Wv, Wvbf, n4w);
    k_cvt    <<<dim3(n4w / 256), dim3(256), 0, stream>>>(Wo, Wobf, n4w);

    // 3) projections (WMMA GEMMs): grid = (N/128, M/64)
    dim3 gg(HID / 128, S_LEN / 64);
    k_gemm_nt<HID><<<gg, dim3(256), 0, stream>>>(Xbf,  Wqbf, S_LEN, HID, 0, Qbf, nullptr, nullptr);
    k_gemm_nt<HID><<<gg, dim3(256), 0, stream>>>(XSbf, Wkbf, S_LEN, HID, 0, Kbf, nullptr, nullptr);
    k_gemm_nt<HID><<<gg, dim3(256), 0, stream>>>(XSbf, Wvbf, S_LEN, HID, 1, Vt,  nullptr, nullptr);

    // 4) flash attention
    k_attn<<<dim3(S_LEN / 16, NHEAD), dim3(256), 0, stream>>>(Qbf, Kbf, Vt, bmask, Cbf);

    // 5) output projection + residual (f32)
    k_gemm_nt<HID><<<gg, dim3(256), 0, stream>>>(Cbf, Wobf, S_LEN, HID, 2, nullptr, Res, inputs);

    // 6) LayerNorm
    k_layernorm<<<dim3(S_LEN), dim3(256), 0, stream>>>(Res, lnw, lnb, out);
}
